// MaskedAttention_4587025072514
// MI455X (gfx1250) — matrix operand order repaired and checked by reading, not compiled, not run on hardware
//
#include <hip/hip_runtime.h>
#include <hip/hip_bf16.h>
#include <stdint.h>

// ---------------------------------------------------------------------------
// MaskedAttention for MI455X (gfx1250, wave32, WMMA)
//   stage 1: qkv_gemm  — Y = X * W^T   (fp32 in -> bf16 out, f32 accum WMMA)
//   stage 2: attn      — flash-attention; Q/K staged via async-to-LDS,
//                        V block via TDM tensor_load_to_lds (TENSORcnt),
//                        scores & P·V on v_wmma_f32_16x16x32_bf16.
// ---------------------------------------------------------------------------

#define HD   1024
#define SEQ  4096
#define NBAT 4

typedef __attribute__((ext_vector_type(16))) __bf16 v16bf;
typedef __attribute__((ext_vector_type(8)))  float  v8f;
typedef __attribute__((ext_vector_type(4)))  unsigned int v4u;
typedef __attribute__((ext_vector_type(8)))  int v8i;
typedef __attribute__((ext_vector_type(4)))  int v4i;

#define AS1 __attribute__((address_space(1)))
#define AS3 __attribute__((address_space(3)))

#if __has_builtin(__builtin_amdgcn_global_load_async_to_lds_b128)
#define HAVE_ASYNC 1
#else
#define HAVE_ASYNC 0
#endif

#if __has_builtin(__builtin_amdgcn_tensor_load_to_lds)
#define HAVE_TDM 1
#else
#define HAVE_TDM 0
#endif

__device__ __forceinline__ void wait_async0() {
#if HAVE_ASYNC
#if __has_builtin(__builtin_amdgcn_s_wait_asynccnt)
  __builtin_amdgcn_s_wait_asynccnt(0);
#else
  asm volatile("s_wait_asynccnt 0x0" ::: "memory");
#endif
#endif
}

__device__ __forceinline__ void wait_tensor0() {
#if __has_builtin(__builtin_amdgcn_s_wait_tensorcnt)
  __builtin_amdgcn_s_wait_tensorcnt(0);
#else
  asm volatile("s_wait_tensorcnt 0x0" ::: "memory");
#endif
}

// 16-byte global->LDS copy: async (ASYNCcnt, bypasses VGPRs) when available.
// Builtin signature (from compiler diagnostic): arg0 = int4 ptr in AS1
// (__device__), arg1 = LDS-side ptr, then (offset, cpol) immediates.
__device__ __forceinline__ void copy16B(__bf16* ldst, const __bf16* gsrc) {
#if HAVE_ASYNC
  unsigned loff = (unsigned)(uintptr_t)ldst;                  // low32 == DS offset
  AS3 v4i* lp   = (AS3 v4i*)(uintptr_t)loff;                  // inttoptr, no AS cast
  AS1 v4i* gp   = (AS1 v4i*)(uintptr_t)gsrc;
  __builtin_amdgcn_global_load_async_to_lds_b128(gp, lp, 0, 0);
#else
  *reinterpret_cast<uint4*>(ldst) = *reinterpret_cast<const uint4*>(gsrc);
#endif
}

#if HAVE_TDM
// TDM: DMA a 2-D tile (rows x row_elems of 2-byte elements, global row stride
// row_stride_elems) into contiguous LDS. D# built per CDNA5 ISA 8.3/8.4.
__device__ __forceinline__ void tdm_load_2d(const void* gsrc, void* ldst,
                                            unsigned rows, unsigned row_elems,
                                            unsigned row_stride_elems) {
  unsigned lds = (unsigned)(uintptr_t)ldst;                   // DS byte offset
  unsigned long long ga = (unsigned long long)(uintptr_t)gsrc;
  v4u g0;
  g0[0] = 1u;                                   // count=1, user descriptor
  g0[1] = lds;                                  // lds_addr
  g0[2] = (unsigned)(ga & 0xffffffffu);         // global_addr[31:0]
  g0[3] = (unsigned)((ga >> 32) & 0x01ffffffu)  // global_addr[56:32]
        | (2u << 30);                           // type = 2 ("image")
  v8i g1;
  g1[0] = (1 << 16);                                            // data_size=2B
  g1[1] = (int)((row_elems & 0xffffu) << 16);                   // tensor_dim0 lo
  g1[2] = (int)((row_elems >> 16) & 0xffffu)                    // tensor_dim0 hi
        | (int)((rows & 0xffffu) << 16);                        // tensor_dim1 lo
  g1[3] = (int)((rows >> 16) & 0xffffu)                         // tensor_dim1 hi
        | (int)((row_elems & 0xffffu) << 16);                   // tile_dim0
  g1[4] = (int)(rows & 0xffffu);                                // tile_dim1
  g1[5] = (int)row_stride_elems;                                // dim0_stride lo
  g1[6] = 0;
  g1[7] = 0;
  v4i g2 = (v4i){0, 0, 0, 0};
  v4i g3 = (v4i){0, 0, 0, 0};
#if defined(__clang_major__) && (__clang_major__ >= 23)
  v8i g4 = (v8i){0, 0, 0, 0, 0, 0, 0, 0};
  __builtin_amdgcn_tensor_load_to_lds(g0, g1, g2, g3, g4, 0);
#else
  __builtin_amdgcn_tensor_load_to_lds(g0, g1, g2, g3, 0);
#endif
}
#endif

__device__ __forceinline__ v8f wmma_bf16(v16bf a, v16bf b, v8f c) {
  return __builtin_amdgcn_wmma_f32_16x16x32_bf16(false, a, false, b,
                                                 (short)0, c, false, false);
}

// A fragment (16x32 bf16) from LDS tile laid out [rows][ldk].
__device__ __forceinline__ v16bf load_a_frag(const __bf16* lds, int ldk,
                                             int row0, int k0, int lane) {
  const __bf16* p = lds + (row0 + (lane & 15)) * ldk + k0;
  const int kb = (lane >> 4) * 16;
  v16bf a;
#pragma unroll
  for (int v = 0; v < 8; ++v) {
    int k = kb + 2 * v;
    a[2 * v]     = p[k];
    a[2 * v + 1] = p[k + 1];
  }
  return a;
}

// B fragment (32x16, KxN) where LDS source holds [N][K].
__device__ __forceinline__ v16bf load_b_nmajor(const __bf16* lds, int ldk,
                                               int n0, int k0, int lane) {
  const __bf16* p = lds + (n0 + (lane & 15)) * ldk + k0 + (lane >> 4) * 16;
  v16bf b;
#pragma unroll
  for (int v = 0; v < 16; ++v) b[v] = p[v];
  return b;
}

// B fragment (32x16, KxN) where LDS source holds [K][N].
__device__ __forceinline__ v16bf load_b_kmajor(const __bf16* lds, int ldn,
                                               int n0, int k0, int lane) {
  const int n  = n0 + (lane & 15);
  const int kb = k0 + (lane >> 4) * 16;
  v16bf b;
#pragma unroll
  for (int v = 0; v < 16; ++v) b[v] = lds[(kb + v) * ldn + n];
  return b;
}

union Pack8 { __bf16 h[8]; uint4 u; };

__device__ __forceinline__ void cvt_store8(__bf16* dst, float4 a, float4 b) {
  Pack8 p;
  p.h[0] = (__bf16)a.x; p.h[1] = (__bf16)a.y; p.h[2] = (__bf16)a.z; p.h[3] = (__bf16)a.w;
  p.h[4] = (__bf16)b.x; p.h[5] = (__bf16)b.y; p.h[6] = (__bf16)b.z; p.h[7] = (__bf16)b.w;
  *reinterpret_cast<uint4*>(dst) = p.u;
}

// ---------------------------------------------------------------------------
// Stage 1: Y[m,n] = sum_k X[m,k] * W[n,k]  (M=16384, N=K=1024), Y in bf16.
// ---------------------------------------------------------------------------
#define GLD 40  // padded LDS K-stride (bf16): 80B = 20 banks, conflict-free

__global__ __launch_bounds__(256) void qkv_gemm(const float* __restrict__ X,
                                                const float* __restrict__ W,
                                                __bf16* __restrict__ Y) {
  __shared__ __align__(16) __bf16 xs[64 * GLD];
  __shared__ __align__(16) __bf16 wsh[128 * GLD];

  const int tid = threadIdx.x, lane = tid & 31, wv = tid >> 5;
  const int m0 = blockIdx.y * 64, n0 = blockIdx.x * 128;
  const int wm = wv & 1, wn = wv >> 1;

  v8f acc[2][2];
#pragma unroll
  for (int i = 0; i < 2; ++i)
#pragma unroll
    for (int j = 0; j < 2; ++j)
      acc[i][j] = (v8f){0.f, 0.f, 0.f, 0.f, 0.f, 0.f, 0.f, 0.f};

  const int xrow = tid >> 2, xcol = (tid & 3) * 8;
  const int wrow = tid >> 1, wcol = (tid & 1) * 16;

  for (int kk = 0; kk < HD; kk += 32) {
    const float4* xp =
        reinterpret_cast<const float4*>(X + (size_t)(m0 + xrow) * HD + kk + xcol);
    cvt_store8(xs + xrow * GLD + xcol, xp[0], xp[1]);
    const float4* wp =
        reinterpret_cast<const float4*>(W + (size_t)(n0 + wrow) * HD + kk + wcol);
    cvt_store8(wsh + wrow * GLD + wcol, wp[0], wp[1]);
    cvt_store8(wsh + wrow * GLD + wcol + 8, wp[2], wp[3]);
    if (kk + 32 < HD) {
      __builtin_prefetch(X + (size_t)(m0 + xrow) * HD + kk + 32 + xcol, 0, 1);
      __builtin_prefetch(W + (size_t)(n0 + wrow) * HD + kk + 32 + wcol, 0, 1);
    }
    __syncthreads();

    v16bf a0 = load_a_frag(xs, GLD, 32 * wm, 0, lane);
    v16bf a1 = load_a_frag(xs, GLD, 32 * wm + 16, 0, lane);
    v16bf b0 = load_b_nmajor(wsh, GLD, 32 * wn, 0, lane);
    v16bf b1 = load_b_nmajor(wsh, GLD, 32 * wn + 16, 0, lane);
    acc[0][0] = wmma_bf16(a0, b0, acc[0][0]);
    acc[0][1] = wmma_bf16(a0, b1, acc[0][1]);
    acc[1][0] = wmma_bf16(a1, b0, acc[1][0]);
    acc[1][1] = wmma_bf16(a1, b1, acc[1][1]);
    __syncthreads();
  }

#pragma unroll
  for (int mi = 0; mi < 2; ++mi)
#pragma unroll
    for (int ni = 0; ni < 2; ++ni)
#pragma unroll
      for (int r = 0; r < 8; ++r) {
        int m = m0 + 32 * wm + 16 * mi + r + 8 * (lane >> 4);
        int n = n0 + 32 * wn + 16 * ni + (lane & 15);
        Y[(size_t)m * HD + n] = (__bf16)acc[mi][ni][r];
      }
}

// ---------------------------------------------------------------------------
// Stage 2: flash attention. One workgroup per (batch, 64-query block).
// 256 threads / 8 waves. Key blocks of 32, D streamed in 128-wide chunks.
// V block DMA'd by TDM at key-block start, overlapping the score phase
// (TENSORcnt is independent of the Q/K ASYNCcnt traffic).
// ---------------------------------------------------------------------------
#define BQ  64
#define BK  32
#define DC  128
#define QLD 136  // padded chunk stride (bf16): 272B, 16B aligned, conflict-free

__global__ __launch_bounds__(256) void attn(const __bf16* __restrict__ Q,
                                            const __bf16* __restrict__ K,
                                            const __bf16* __restrict__ V,
                                            float* __restrict__ O) {
  __shared__ __align__(16) __bf16 Qs[BQ * QLD];  // 64 x 128 Q chunk
  __shared__ __align__(16) __bf16 Ks[BK * QLD];  // 32 x 128 K chunk
  __shared__ __align__(16) __bf16 Vs[BK * HD];   // 32 x 1024 V block
  __shared__ __align__(16) float  Ss[BQ * BK];   // score tile
  __shared__ __align__(16) __bf16 Ps[BQ * BK];   // softmaxed probs (bf16)
  __shared__ float rowm[BQ], rowl[BQ], rowa[BQ];

  const int tid = threadIdx.x, lane = tid & 31, wv = tid >> 5;
  const int q0 = blockIdx.x * BQ, b = blockIdx.y;
  const int sm = wv >> 1, sn = wv & 1;

  v8f acc[4][8];
#pragma unroll
  for (int mi = 0; mi < 4; ++mi)
#pragma unroll
    for (int ni = 0; ni < 8; ++ni)
      acc[mi][ni] = (v8f){0.f, 0.f, 0.f, 0.f, 0.f, 0.f, 0.f, 0.f};

  if (tid < BQ) { rowm[tid] = -3.0e38f; rowl[tid] = 0.0f; }

  const __bf16* Qg = Q + (size_t)(b * SEQ + q0) * HD;

  for (int k0 = 0; k0 < q0 + BQ; k0 += BK) {  // causal: keys <= q0+BQ-1
    const __bf16* Kg = K + (size_t)(b * SEQ + k0) * HD;
    const __bf16* Vg = V + (size_t)(b * SEQ + k0) * HD;

#if HAVE_TDM
    // Kick off V block DMA now; it overlaps the whole score phase below.
    if (wv == 0) tdm_load_2d(Vg, Vs, BK, HD, HD);
#endif

    // ---- scores: S = Q * K^T over D chunks --------------------------------
    v8f sacc = (v8f){0.f, 0.f, 0.f, 0.f, 0.f, 0.f, 0.f, 0.f};
    for (int dc = 0; dc < HD; dc += DC) {
      {  // Q chunk 64x128 -> LDS (4 x 16B per thread)
        int row = tid >> 2, seg = (tid & 3) * 32;
        const __bf16* src = Qg + (size_t)row * HD + dc + seg;
        __bf16* dst = Qs + row * QLD + seg;
#pragma unroll
        for (int i = 0; i < 4; ++i) copy16B(dst + i * 8, src + i * 8);
      }
      {  // K chunk 32x128 -> LDS (2 x 16B per thread)
        int row = tid >> 3, seg = (tid & 7) * 16;
        const __bf16* src = Kg + (size_t)row * HD + dc + seg;
        __bf16* dst = Ks + row * QLD + seg;
        copy16B(dst, src);
        copy16B(dst + 8, src + 8);
      }
      wait_async0();
      __syncthreads();
#pragma unroll
      for (int ks = 0; ks < DC; ks += 32) {
        v16bf a  = load_a_frag(Qs, QLD, 16 * sm, ks, lane);
        v16bf bb = load_b_nmajor(Ks, QLD, 16 * sn, ks, lane);
        sacc = wmma_bf16(a, bb, sacc);
      }
      __syncthreads();
    }

    // ---- write masked, scaled scores to LDS -------------------------------
#pragma unroll
    for (int r = 0; r < 8; ++r) {
      int m = 16 * sm + r + 8 * (lane >> 4);
      int c = 16 * sn + (lane & 15);
      float s = sacc[r] * 0.03125f;           // 1/sqrt(1024)
      if (k0 + c > q0 + m) s = -3.0e38f;      // causal mask
      Ss[m * BK + c] = s;
    }

#if HAVE_TDM
    if (wv == 0) wait_tensor0();              // V DMA must be in LDS now
#else
    {  // V block 32x1024 -> LDS (16 x 16B per thread)
      int key = tid >> 3, seg = (tid & 7) * 128;
      const __bf16* src = Vg + (size_t)key * HD + seg;
      __bf16* dst = Vs + key * HD + seg;
#pragma unroll
      for (int i = 0; i < 16; ++i) copy16B(dst + i * 8, src + i * 8);
    }
    wait_async0();
#endif
    __syncthreads();

    // ---- online softmax: one query row per thread -------------------------
    if (tid < BQ) {
      int m = tid;
      float mo = rowm[m], bm = mo;
      for (int c = 0; c < BK; ++c) bm = fmaxf(bm, Ss[m * BK + c]);
      float al = __expf(mo - bm);
      float l  = rowl[m] * al;
      for (int c = 0; c < BK; ++c) {
        float p = __expf(Ss[m * BK + c] - bm);
        Ps[m * BK + c] = (__bf16)p;
        l += p;
      }
      rowm[m] = bm; rowl[m] = l; rowa[m] = al;
    }
    __syncthreads();

    // ---- rescale accumulators by exp(m_old - m_new) -----------------------
#pragma unroll
    for (int mi = 0; mi < 4; ++mi)
#pragma unroll
      for (int r = 0; r < 8; ++r) {
        float f = rowa[16 * mi + r + 8 * (lane >> 4)];
#pragma unroll
        for (int ni = 0; ni < 8; ++ni) acc[mi][ni][r] *= f;
      }

    // ---- O += P * V --------------------------------------------------------
    v16bf af[4];
#pragma unroll
    for (int mi = 0; mi < 4; ++mi) af[mi] = load_a_frag(Ps, BK, 16 * mi, 0, lane);
#pragma unroll
    for (int ni = 0; ni < 8; ++ni) {
      v16bf bb = load_b_kmajor(Vs, HD, 128 * wv + 16 * ni, 0, lane);
#pragma unroll
      for (int mi = 0; mi < 4; ++mi) acc[mi][ni] = wmma_bf16(af[mi], bb, acc[mi][ni]);
    }
    __syncthreads();
  }

  // ---- normalize and store fp32 output ------------------------------------
  float* Og = O + (size_t)(b * SEQ + q0) * HD;
#pragma unroll
  for (int mi = 0; mi < 4; ++mi)
#pragma unroll
    for (int r = 0; r < 8; ++r) {
      int m = 16 * mi + r + 8 * (lane >> 4);
      float inv = 1.0f / rowl[m];
#pragma unroll
      for (int ni = 0; ni < 8; ++ni) {
        int n = 128 * wv + 16 * ni + (lane & 15);
        Og[(size_t)m * HD + n] = acc[mi][ni][r] * inv;
      }
    }
}

// ---------------------------------------------------------------------------
extern "C" void kernel_launch(void* const* d_in, const int* in_sizes, int n_in,
                              void* d_out, int out_size, void* d_ws, size_t ws_size,
                              hipStream_t stream) {
  (void)in_sizes; (void)n_in; (void)out_size; (void)ws_size;
  const float* x  = (const float*)d_in[0];
  const float* Wq = (const float*)d_in[1];
  const float* Wk = (const float*)d_in[2];
  const float* Wv = (const float*)d_in[3];
  float* out = (float*)d_out;

  const size_t nElem = (size_t)NBAT * SEQ * HD;  // 16.7M elems, 32MB as bf16
  __bf16* Qb = (__bf16*)d_ws;
  __bf16* Kb = Qb + nElem;
  __bf16* Vb = Kb + nElem;

  dim3 gg(HD / 128, (NBAT * SEQ) / 64);  // (8, 256)
  qkv_gemm<<<gg, 256, 0, stream>>>(x, Wq, Qb);
  qkv_gemm<<<gg, 256, 0, stream>>>(x, Wk, Kb);
  qkv_gemm<<<gg, 256, 0, stream>>>(x, Wv, Vb);

  dim3 ga(SEQ / BQ, NBAT);  // (64, 4)
  attn<<<ga, 256, 0, stream>>>(Qb, Kb, Vb, out);
}
